// MultiLatentDecoder_7576322310199
// MI455X (gfx1250) — matrix operand order repaired and checked by reading, not compiled, not run on hardware
//
#include <hip/hip_runtime.h>
#include <hip/hip_bf16.h>
#include <cstddef>

// ---------------------------------------------------------------------------
// MultiLatentDecoder forward on gfx1250 (MI455X), wave32 + WMMA bf16.
//
// Forward reduces to: per entry n, d = argmax_d alpha[d,n];
//   out[n] = relu(relu(round(w[n])/div @ S0[d] + b0[d]) @ S1[d] + b1[d]) @ S2[d] + b2[d]
//
// Pipeline (all on `stream`, graph-capture safe):
//   K0: zero 8 per-decoder counters in workspace
//   K1: convert all decoder weights f32 -> bf16 into workspace
//   K2: bucket via per-block LDS histogram + one global atomic per (block,decoder)
//   K3: grouped 3-layer MLP per decoder segment with v_wmma_f32_16x16x32_bf16
//
// Workspace layout (needs 256 + 8*N*4 + 8*26624*2 bytes ~= 2.5 MB):
//   [0,32)                  : int counts[8]
//   [256, 256+8N*4)         : int perm[8][N]
//   [.., +8*26624*2)        : bf16 weights, per decoder: W0[64x128] W1[128x128] W2[128x16]
// ---------------------------------------------------------------------------

typedef __bf16 bf16_t;
typedef __attribute__((ext_vector_type(16))) __bf16 bf16x16;
typedef __attribute__((ext_vector_type(8)))  float  f32x8;

#define NDEC   8
#define LAT    64
#define HID    128
#define FEAT   16
#define WPER   (LAT*HID + HID*HID + HID*FEAT)   // 26624 elems per decoder
#define HSTRIDE 132                              // padded row stride (floats): bank = (4r+c)%64, conflict-free

static __device__ __forceinline__ f32x8 wmma_bf16(bf16x16 a, bf16x16 b, f32x8 c) {
    // (neg_a, A, neg_b, B, c_mod, C, reuse_a, reuse_b)
    return __builtin_amdgcn_wmma_f32_16x16x32_bf16(false, a, false, b, (short)0, c, false, false);
}

__global__ void mld_zero_counts(int* counts) {
    if (threadIdx.x < NDEC) counts[threadIdx.x] = 0;
}

__global__ void mld_convert_weights(const float* __restrict__ s0,
                                    const float* __restrict__ s1,
                                    const float* __restrict__ s2,
                                    bf16_t* __restrict__ wbf, int total) {
    int t = blockIdx.x * blockDim.x + threadIdx.x;
    if (t >= total) return;
    int d = t / WPER;
    int r = t - d * WPER;
    float v;
    if (r < LAT * HID)                  v = s0[d * (LAT * HID) + r];
    else if (r < LAT * HID + HID * HID) v = s1[d * (HID * HID) + (r - LAT * HID)];
    else                                v = s2[d * (HID * FEAT) + (r - LAT * HID - HID * HID)];
    wbf[t] = (bf16_t)v;
}

// Two-level bucketing: LDS histogram (ds_add_rtn_u32) -> 8 global atomics per
// block to reserve segment bases -> conflict-free index scatter. Avoids 65536
// serialized global RMWs on 8 hot addresses.
__global__ void mld_bucket(const float* __restrict__ alpha,
                           int* __restrict__ counts,
                           int* __restrict__ perm, int N) {
    __shared__ int lcnt[NDEC];
    __shared__ int lbase[NDEC];
    const int tid = threadIdx.x;
    if (tid < NDEC) lcnt[tid] = 0;
    __syncthreads();

    const int n = blockIdx.x * blockDim.x + tid;
    int bi = 0, pos = 0;
    const bool active = (n < N);
    if (active) {
        float best = alpha[n];
#pragma unroll
        for (int d = 1; d < NDEC; ++d) {
            float v = alpha[(size_t)d * N + n];
            if (v > best) { best = v; bi = d; }   // strict '>' => first max, matches argmax
        }
        pos = atomicAdd(&lcnt[bi], 1);            // LDS atomic, per-block
    }
    __syncthreads();
    if (tid < NDEC) lbase[tid] = (lcnt[tid] > 0) ? atomicAdd(&counts[tid], lcnt[tid]) : 0;
    __syncthreads();
    if (active) perm[(size_t)bi * N + lbase[bi] + pos] = n;
}

// 4 waves / block, one 16-row tile per wave => 64 entries per block.
__global__ __launch_bounds__(128, 1)
void mld_decode(const float* __restrict__ weight,
                const float* __restrict__ divv,
                const int*   __restrict__ perm,
                const int*   __restrict__ counts,
                const bf16_t* __restrict__ wbf,
                const float* __restrict__ shift0,
                const float* __restrict__ shift1,
                const float* __restrict__ shift2,
                float* __restrict__ out,
                int N, int bpd) {
    __shared__ float lds_h[4 * 16 * HSTRIDE];   // per-wave 16x128 activation scratch (padded)

    const int d  = blockIdx.x / bpd;
    const int tb = blockIdx.x - d * bpd;
    const int cnt = counts[d];
    const int blockRow = tb * 64;
    if (blockRow >= cnt) return;                 // block-uniform

    const int tid  = threadIdx.x;
    const int wave = tid >> 5;
    const int lane = tid & 31;
    const int rowBase = blockRow + wave * 16;
    if (rowBase >= cnt) return;                  // wave-uniform: EXEC stays all-ones for WMMA

    const int r16 = lane & 15;
    const int g   = lane >> 4;

    const int*    permd = perm + (size_t)d * N;
    const bf16_t* wd    = wbf + (size_t)d * WPER;
    float*        lh    = lds_h + wave * (16 * HSTRIDE);

    __builtin_prefetch(wd, 0, 1);                // global_prefetch_b8: pull weight slab toward caches
    __builtin_prefetch(wd + LAT * HID, 0, 1);

    // Gather this lane's activation row (clamped for tail tiles; dup rows are never stored).
    int myRow = rowBase + r16;
    if (myRow >= cnt) myRow = cnt - 1;
    const int nIdx = permd[myRow];
    const float* xrow = weight + (size_t)nIdx * LAT;

    // ---- A0 fragments: x = round(weight)/div as 2 chunks of 16x32 bf16.
    // A layout (16-bit, 16x32): lane(0-15)=row M, half g: K in [g*8,g*8+8) U [16+g*8,16+g*8+8).
    bf16x16 a0[2];
#pragma unroll
    for (int kc = 0; kc < 2; ++kc) {
        const int k0 = kc * 32 + g * 8;
        float4 f0 = *(const float4*)(xrow + k0);
        float4 f1 = *(const float4*)(xrow + k0 + 4);
        float4 f2 = *(const float4*)(xrow + k0 + 16);
        float4 f3 = *(const float4*)(xrow + k0 + 20);
        float xv[16] = {f0.x, f0.y, f0.z, f0.w, f1.x, f1.y, f1.z, f1.w,
                        f2.x, f2.y, f2.z, f2.w, f3.x, f3.y, f3.z, f3.w};
#pragma unroll
        for (int j = 0; j < 8; ++j) {
            a0[kc][j]     = (bf16_t)(__builtin_rintf(xv[j])     / divv[k0 + j]);
            a0[kc][8 + j] = (bf16_t)(__builtin_rintf(xv[8 + j]) / divv[k0 + 16 + j]);
        }
    }

    // ---- Layer 0: [16x64] @ W0[64x128] + shift0 -> relu -> LDS
#pragma unroll
    for (int jo = 0; jo < HID / 16; ++jo) {
        const float s = shift0[d * HID + jo * 16 + r16];
        f32x8 acc = {s, s, s, s, s, s, s, s};
#pragma unroll
        for (int kc = 0; kc < 2; ++kc) {
            // B layout (16-bit, 32x16): lane(0-15)=col N, half g: K in [g*8,g*8+8) U [16+g*8,16+g*8+8)
            bf16x16 b;
#pragma unroll
            for (int j = 0; j < 8; ++j) {
                b[j]     = wd[(size_t)(kc * 32 + g * 8 + j) * HID + jo * 16 + r16];
                b[8 + j] = wd[(size_t)(kc * 32 + 16 + g * 8 + j) * HID + jo * 16 + r16];
            }
            acc = wmma_bf16(a0[kc], b, acc);
        }
#pragma unroll
        for (int v = 0; v < 8; ++v) {
            float hv = acc[v] > 0.f ? acc[v] : 0.f;     // relu
            lh[(v + 8 * g) * HSTRIDE + jo * 16 + r16] = hv;
        }
    }
    asm volatile("s_wait_dscnt 0x0" ::: "memory");

    // ---- Layer 1: [16x128] @ W1[128x128] + shift1 -> relu -> LDS
    bf16x16 a1[4];
#pragma unroll
    for (int kc = 0; kc < 4; ++kc) {
#pragma unroll
        for (int j = 0; j < 8; ++j) {
            a1[kc][j]     = (bf16_t)lh[r16 * HSTRIDE + kc * 32 + g * 8 + j];
            a1[kc][8 + j] = (bf16_t)lh[r16 * HSTRIDE + kc * 32 + 16 + g * 8 + j];
        }
    }
    const bf16_t* w1 = wd + LAT * HID;
#pragma unroll
    for (int jo = 0; jo < HID / 16; ++jo) {
        const float s = shift1[d * HID + jo * 16 + r16];
        f32x8 acc = {s, s, s, s, s, s, s, s};
#pragma unroll
        for (int kc = 0; kc < 4; ++kc) {
            bf16x16 b;
#pragma unroll
            for (int j = 0; j < 8; ++j) {
                b[j]     = w1[(size_t)(kc * 32 + g * 8 + j) * HID + jo * 16 + r16];
                b[8 + j] = w1[(size_t)(kc * 32 + 16 + g * 8 + j) * HID + jo * 16 + r16];
            }
            acc = wmma_bf16(a1[kc], b, acc);
        }
#pragma unroll
        for (int v = 0; v < 8; ++v) {
            float hv = acc[v] > 0.f ? acc[v] : 0.f;
            lh[(v + 8 * g) * HSTRIDE + jo * 16 + r16] = hv;
        }
    }
    asm volatile("s_wait_dscnt 0x0" ::: "memory");

    // ---- Layer 2: [16x128] @ W2[128x16] + shift2 (no activation) -> scatter store
    bf16x16 a2[4];
#pragma unroll
    for (int kc = 0; kc < 4; ++kc) {
#pragma unroll
        for (int j = 0; j < 8; ++j) {
            a2[kc][j]     = (bf16_t)lh[r16 * HSTRIDE + kc * 32 + g * 8 + j];
            a2[kc][8 + j] = (bf16_t)lh[r16 * HSTRIDE + kc * 32 + 16 + g * 8 + j];
        }
    }
    const bf16_t* w2 = wd + LAT * HID + HID * HID;
    {
        const float s = shift2[d * FEAT + r16];
        f32x8 acc = {s, s, s, s, s, s, s, s};
#pragma unroll
        for (int kc = 0; kc < 4; ++kc) {
            bf16x16 b;
#pragma unroll
            for (int j = 0; j < 8; ++j) {
                b[j]     = w2[(size_t)(kc * 32 + g * 8 + j) * FEAT + r16];
                b[8 + j] = w2[(size_t)(kc * 32 + 16 + g * 8 + j) * FEAT + r16];
            }
            acc = wmma_bf16(a2[kc], b, acc);
        }
#pragma unroll
        for (int v = 0; v < 8; ++v) {
            const int row = rowBase + v + 8 * g;
            if (row < cnt) {
                const int n2 = permd[row];
                out[(size_t)n2 * FEAT + r16] = acc[v];
            }
        }
    }
}

extern "C" void kernel_launch(void* const* d_in, const int* in_sizes, int n_in,
                              void* d_out, int out_size, void* d_ws, size_t ws_size,
                              hipStream_t stream) {
    const float* weight = (const float*)d_in[0];   // [N, 64]
    const float* alpha  = (const float*)d_in[1];   // [8, N]
    const float* divv   = (const float*)d_in[2];   // [64]
    const float* scale0 = (const float*)d_in[3];   // [8, 64, 128]
    const float* shift0 = (const float*)d_in[4];   // [8, 1, 128]
    const float* scale1 = (const float*)d_in[5];   // [8, 128, 128]
    const float* shift1 = (const float*)d_in[6];   // [8, 1, 128]
    const float* scale2 = (const float*)d_in[7];   // [8, 128, 16]
    const float* shift2 = (const float*)d_in[8];   // [8, 1, 16]
    float* out = (float*)d_out;                    // [N, 16]

    const int N = in_sizes[0] / LAT;

    char* ws = (char*)d_ws;
    int*    counts = (int*)ws;
    int*    perm   = (int*)(ws + 256);
    bf16_t* wbf    = (bf16_t*)(ws + 256 + (size_t)NDEC * N * sizeof(int));

    mld_zero_counts<<<1, 32, 0, stream>>>(counts);

    const int totalW = NDEC * WPER;
    mld_convert_weights<<<(totalW + 255) / 256, 256, 0, stream>>>(scale0, scale1, scale2, wbf, totalW);

    mld_bucket<<<(N + 255) / 256, 256, 0, stream>>>(alpha, counts, perm, N);

    const int bpd = (N + 63) / 64;   // 64 rows (4 tiles) per block
    mld_decode<<<NDEC * bpd, 128, 0, stream>>>(weight, divv, perm, counts, wbf,
                                               shift0, shift1, shift2, out, N, bpd);
}
